// BConv_69561290326218
// MI455X (gfx1250) — matrix operand order repaired and checked by reading, not compiled, not run on hardware
//
#include <hip/hip_runtime.h>
#include <hip/hip_bf16.h>

// ---------------------------------------------------------------------------
// Binary-weight 3x3 conv (stride 1, pad 1) as implicit GEMM on CDNA5 WMMA.
//
//   out[32,256,56,56] = conv(x[32,256,56,56], sign(W[256,256,3,3]))
//
// Roofline: 118.4 GFLOP; x+out+W ~208 MB fits 192MB L2 -> HBM floor ~9us;
// bf16 WMMA (16x16x32, f32 acc) ~47-53us -> WMMA-issue bound, with A-tile L2
// reads the secondary limiter. Weights (+-1) are exact in bf16.
//
// v3 vs v2:
//  * WG tile = 128 Cout x 4 output rows (256 cols): A L2 traffic 2.06GB->0.5GB.
//  * Whole-chunk A (9 taps, 72KB) async-loaded to LDS once per cin-chunk
//    (global_load_async_to_lds_b128, ASYNCcnt) -> tap loop barrier-free,
//    fully unrolled: 72 WMMA between barriers, 2 barriers/chunk.
//  * Branch-free Xs staging: clamped-address loads + cndmask halo zeroing.
//  * 153.6KB dynamic LDS; 512 thr = 16 wave32 (2M x 8N), 64x32 per wave.
// ---------------------------------------------------------------------------

typedef __bf16 bf16_t;
typedef __attribute__((ext_vector_type(16))) __bf16 v16bf;
typedef __attribute__((ext_vector_type(8)))  __bf16 v8bf;
typedef __attribute__((ext_vector_type(8)))  float  v8f;

#define CIN   256
#define COUT  256
#define HW    56
#define IMGN  32
#define KTAPS 9
#define KC    32            // K per WMMA step (channels per chunk)
#define NCHUNK (CIN / KC)   // 8
#define MT    128           // Cout tile per WG
#define ROWS  4             // output rows per WG (N-tile = 4*64 = 256 cols)
#define XROWS (ROWS + 2)    // staged input rows
#define LDA   40            // LDS K-stride in bf16 (80B: 16B-aligned, bank-safe)

#define XS_BYTES   (XROWS * 64 * LDA * 2)      // 30720 B per buffer
#define XS_OFF(b)  ((b) * XS_BYTES)
#define AS_TAP     (MT * LDA * 2)              // 10240 B per tap
#define AS_OFF     (2 * XS_BYTES)              // 61440
#define SMEM_BYTES (AS_OFF + KTAPS * AS_TAP)   // 153600 B (dynamic LDS)

__device__ __forceinline__ bf16_t f2bf(float f) {
    union { float f; unsigned u; } c; c.f = f;
    unsigned u = c.u + 0x7FFFu + ((c.u >> 16) & 1u);   // RNE
    union { unsigned short s; bf16_t b; } r; r.s = (unsigned short)(u >> 16);
    return r.b;
}

// ---- pack sign(W) as bf16: wpack[tap][cout][cin] --------------------------
__global__ void bconv_pack_w(const float* __restrict__ W,
                             bf16_t* __restrict__ wpack) {
    int linear = blockIdx.x * blockDim.x + threadIdx.x;  // over 9*256*256
    int ci  = linear & (CIN - 1);
    int rest = linear >> 8;
    int m   = rest & (COUT - 1);
    int tap = rest >> 8;
    if (tap >= KTAPS) return;
    float w = W[((size_t)m * CIN + ci) * KTAPS + tap];   // [m][ci][ky][kx]
    float s = (w > 0.f) ? 1.f : ((w < 0.f) ? -1.f : 0.f);
    wpack[((size_t)tap * COUT + m) * CIN + ci] = f2bf(s);
}

// ---- main implicit-GEMM conv kernel ---------------------------------------
__global__ void __launch_bounds__(512)
bconv_wmma(const float* __restrict__ x,
           const bf16_t* __restrict__ wpack,
           float* __restrict__ out) {
    extern __shared__ unsigned char smem[];

    const int tid  = threadIdx.x;
    const int y0   = blockIdx.x * ROWS;   // output row base (0,4,..,52)
    const int img  = blockIdx.y;          // image 0..31
    const int m0   = blockIdx.z * MT;     // Cout tile base (0 or 128)

    const int lane  = tid & 31;
    const int wv    = tid >> 5;           // wave 0..15
    const int waveM = wv >> 3;            // 0..1 -> 64-row Cout block
    const int waveN = wv & 7;             // 0..7 -> 32-column block
    const int mrow  = lane & 15;
    const bool hiL  = (lane >= 16);

    const unsigned lds_base = (unsigned)(size_t)(void*)smem;
    const unsigned long long wbase = (unsigned long long)(const void*)wpack;

    v8f acc[4][2];
#pragma unroll
    for (int i = 0; i < 4; ++i)
#pragma unroll
        for (int j = 0; j < 2; ++j) acc[i][j] = {};

    // ---- Xs: activation patch [xrow(6)][slot64][ch, LDA], slot s = pixel s-1.
    // Branch-free: clamped-address load + select-0 for halo/out-of-image.
    auto load_xs = [&](int chunk, int xb) {
        const int ci0 = chunk * KC;
        bf16_t* xs = (bf16_t*)(smem + XS_OFF(xb));
#pragma unroll 6
        for (int it = 0; it < (XROWS * 64 * KC) / 512; ++it) {  // 24
            int lin  = it * 512 + tid;
            int slot = lin & 63;                 // fast: coalesced px
            int ch   = (lin >> 6) & 31;
            int row  = lin >> 11;                // 0..5
            int yy   = y0 + row - 1;
            int px   = slot - 1;
            int yyc  = min(max(yy, 0), HW - 1);
            int pxc  = min(max(px, 0), HW - 1);
            float v = x[((size_t)img * CIN + ci0 + ch) * (HW * HW)
                        + (size_t)yyc * HW + pxc];
            bool ok = ((unsigned)yy < (unsigned)HW) & ((unsigned)px < (unsigned)HW);
            v = ok ? v : 0.f;
            xs[(row * 64 + slot) * LDA + ch] = f2bf(v);
        }
    };

    // ---- A: whole chunk (9 taps x 128 x 32 bf16) async global->LDS --------
    auto load_a_async = [&](int chunk) {
        const int ci0 = chunk * KC;
        const int m  = tid >> 2;                 // 0..127
        const int kq = tid & 3;                  // 8-bf16 quad
        const unsigned lds0  = lds_base + AS_OFF + (unsigned)(m * LDA + kq * 8) * 2u;
        const unsigned goff0 = (unsigned)(((m0 + m) * CIN + ci0 + kq * 8) * 2);
#pragma unroll
        for (int tap = 0; tap < KTAPS; ++tap) {
            unsigned lds  = lds0  + (unsigned)(tap * AS_TAP);
            unsigned goff = goff0 + (unsigned)(tap * COUT * CIN * 2);
            asm volatile("global_load_async_to_lds_b128 %0, %1, %2"
                         :: "v"(lds), "v"(goff), "s"(wbase) : "memory");
        }
    };

    // ---- prologue ----
    load_xs(0, 0);
    load_a_async(0);
    asm volatile("s_wait_asynccnt 0" ::: "memory");
    __syncthreads();

    int xb = 0;
    for (int chunk = 0; chunk < NCHUNK; ++chunk) {
        if (chunk + 1 < NCHUNK) load_xs(chunk + 1, xb ^ 1);
        const bf16_t* xs = (const bf16_t*)(smem + XS_OFF(xb));

#pragma unroll
        for (int tap = 0; tap < KTAPS; ++tap) {
            const bf16_t* as_ = (const bf16_t*)(smem + AS_OFF + tap * AS_TAP);
            const int ky = tap / 3, kx = tap % 3;

            // A frags: lanes<16 hold K{0-7,16-23}, lanes>=16 K{8-15,24-31}
            v16bf afr[4];
#pragma unroll
            for (int i = 0; i < 4; ++i) {
                const bf16_t* arow = as_ + (waveM * 64 + i * 16 + mrow) * LDA;
                v8bf lo = *(const v8bf*)(arow + (hiL ? 8 : 0));
                v8bf hi = *(const v8bf*)(arow + (hiL ? 24 : 16));
                afr[i] = __builtin_shufflevector(lo, hi,
                    0,1,2,3,4,5,6,7,8,9,10,11,12,13,14,15);
            }
            // B frags: column n = rowsel*64+px -> Xs row rowsel+ky, slot px+kx.
            v16bf bfr[2];
#pragma unroll
            for (int j = 0; j < 2; ++j) {
                int n = waveN * 32 + j * 16 + mrow;
                int rowsel = n >> 6, px = n & 63;
                const bf16_t* bcol =
                    xs + ((rowsel + ky) * 64 + px + kx) * LDA + (hiL ? 8 : 0);
                v8bf lo = *(const v8bf*)(bcol);
                v8bf hi = *(const v8bf*)(bcol + 16);
                bfr[j] = __builtin_shufflevector(lo, hi,
                    0,1,2,3,4,5,6,7,8,9,10,11,12,13,14,15);
            }
#pragma unroll
            for (int i = 0; i < 4; ++i)
#pragma unroll
                for (int j = 0; j < 2; ++j)
                    acc[i][j] = __builtin_amdgcn_wmma_f32_16x16x32_bf16(
                        false, afr[i], false, bfr[j],
                        (short)0, acc[i][j], false, false);
        }

        __syncthreads();                       // A reads + Xs[xb^1] stores done
        if (chunk + 1 < NCHUNK) {
            load_a_async(chunk + 1);           // overwrite A region
            asm volatile("s_wait_asynccnt 0" ::: "memory");
        }
        __syncthreads();                       // A writes visible to all waves
        xb ^= 1;
    }

    // ---- store: C/D layout VGPR r -> M = r + (lane<16 ? 0 : 8), N = lane%16
    const int mh = hiL ? 8 : 0;
#pragma unroll
    for (int j = 0; j < 2; ++j) {
        int n = waveN * 32 + j * 16 + mrow;
        int rowsel = n >> 6, px = n & 63;
        if (px >= HW) continue;
        int yrow = y0 + rowsel;
#pragma unroll
        for (int i = 0; i < 4; ++i) {
            int cobase = m0 + waveM * 64 + i * 16 + mh;
            float* op = out + (((size_t)img * COUT + cobase) * HW + yrow) * HW + px;
#pragma unroll
            for (int r = 0; r < 8; ++r)
                op[(size_t)r * (HW * HW)] = acc[i][j][r];
        }
    }
}

extern "C" void kernel_launch(void* const* d_in, const int* in_sizes, int n_in,
                              void* d_out, int out_size, void* d_ws, size_t ws_size,
                              hipStream_t stream) {
    (void)in_sizes; (void)n_in; (void)out_size; (void)ws_size;
    const float* x = (const float*)d_in[0];
    const float* W = (const float*)d_in[1];
    float* out     = (float*)d_out;
    bf16_t* wpack  = (bf16_t*)d_ws;   // 9*256*256*2 = 1.13 MB scratch

    bconv_pack_w<<<(KTAPS * COUT * CIN) / 256, 256, 0, stream>>>(W, wpack);

    dim3 grid(HW / ROWS, IMGN, COUT / MT);   // (row-quad, image, cout-half)
    bconv_wmma<<<grid, 512, SMEM_BYTES, stream>>>(x, wpack, out);
}
